// LoHAWrapper_56581899157528
// MI455X (gfx1250) — compile-verified, matrix operand order checked by reading, not run on hardware
//
#include <hip/hip_runtime.h>
#include <hip/hip_bf16.h>

// ---------------------------------------------------------------------------
// LoHA forward for MI455X (gfx1250):
//   Keff = W + (s*a0@b0) .* (s*a1@b1),  out = x @ Keff + bias
// Compute-bound 16384x4096x4096 GEMM -> v_wmma_f32_16x16x32_bf16 with
// 3-product bf16 hi/lo split (~fp32 accuracy at 3/8 the cost of f32 WMMA).
// This round: double-buffered LDS + GLOBAL_LOAD_ASYNC_TO_LDS_B128 for the
// Keff tiles (ASYNCcnt path), software-pipelined x-tile loads for the A side.
// ---------------------------------------------------------------------------

typedef __bf16 bf16_t;
typedef __bf16 v8bf  __attribute__((ext_vector_type(8)));
typedef __bf16 v16bf __attribute__((ext_vector_type(16)));
typedef float  v8f   __attribute__((ext_vector_type(8)));
typedef unsigned int v4u __attribute__((ext_vector_type(4)));
typedef unsigned int v2u __attribute__((ext_vector_type(2)));

#define DDIM 4096
#define RANK 16
#define BM 128
#define BN 128
#define BK 32

// Async global -> LDS copy, 16 bytes per lane, tracked on ASYNCcnt.
__device__ __forceinline__ void async_copy_b128(unsigned lds_off, const void* gptr) {
  asm volatile("global_load_async_to_lds_b128 %0, %1, off"
               :: "v"(lds_off), "v"(gptr)
               : "memory");
}
__device__ __forceinline__ void wait_asynccnt0() {
  asm volatile("s_wait_asynccnt 0x0" ::: "memory");
}

// ---------------------------------------------------------------------------
// Prep: Keff[i][j] = W[i][j] + (s*a0[i,:]@b0[:,j]) * (s*a1[i,:]@b1[:,j])
// Written transposed ([j][i], K-contiguous) and split into bf16 hi/lo so the
// GEMM's B tiles are direct 16-byte async copies into LDS.
// ---------------------------------------------------------------------------
__global__ __launch_bounds__(256)
void loha_prep(const float* __restrict__ W,
               const float* __restrict__ a0, const float* __restrict__ b0,
               const float* __restrict__ a1, const float* __restrict__ b1,
               bf16_t* __restrict__ Bh, bf16_t* __restrict__ Bl) {
  const int j = blockIdx.x * blockDim.x + threadIdx.x;   // d_out
  const int i = blockIdx.y;                              // d_in
  const float s = 1.0f / (float)RANK;                    // ALPHA / RANK
  float p0 = 0.0f, p1 = 0.0f;
#pragma unroll
  for (int r = 0; r < RANK; ++r) {
    p0 += a0[i * RANK + r] * b0[(size_t)r * DDIM + j];
    p1 += a1[i * RANK + r] * b1[(size_t)r * DDIM + j];
  }
  const float keff = W[(size_t)i * DDIM + j] + (s * p0) * (s * p1);
  const bf16_t hi = (bf16_t)keff;
  const float  lo = keff - (float)hi;
  Bh[(size_t)j * DDIM + i] = hi;
  Bl[(size_t)j * DDIM + i] = (bf16_t)lo;
}

// ---------------------------------------------------------------------------
// GEMM: out[M][N=4096] = x[M][K=4096] @ Keff[K][N] + bias
//   block tile 128x128, BK=32, 256 threads = 8 waves (wave32)
//   wave tile 64x32 = 4x2 subtiles; 3 bf16 WMMAs each -> 24 WMMAs / K-step
//   double-buffered LDS; B tiles via async-to-LDS, A tiles via reg pipeline
// ---------------------------------------------------------------------------
__global__ __launch_bounds__(256)
void loha_gemm(const float* __restrict__ X,
               const bf16_t* __restrict__ Bh, const bf16_t* __restrict__ Bl,
               const float* __restrict__ bias,
               float* __restrict__ out) {
  __shared__ bf16_t Ah[2][BM * BK];   // x tile hi, [m][k]
  __shared__ bf16_t Al[2][BM * BK];   // x tile lo residual
  __shared__ bf16_t Bhs[2][BN * BK];  // Keff tile hi, [n][k]
  __shared__ bf16_t Bls[2][BN * BK];  // Keff tile lo

  const int tid   = threadIdx.x;
  const int lane  = tid & 31;
  const int wave  = tid >> 5;
  const int waveM = wave >> 2;       // 0..1 -> 64-row band
  const int waveN = wave & 3;        // 0..3 -> 32-col band
  const int half  = lane >> 4;       // 0: K0-7/16-23, 1: K8-15/24-31
  const int l16   = lane & 15;

  const int m0 = blockIdx.y * BM;
  const int n0 = blockIdx.x * BN;

  v8f acc[4][2];
#pragma unroll
  for (int ms = 0; ms < 4; ++ms)
#pragma unroll
    for (int ns = 0; ns < 2; ++ns)
      acc[ms][ns] = (v8f){0.f, 0.f, 0.f, 0.f, 0.f, 0.f, 0.f, 0.f};

  // per-thread staging coordinates
  const int arow = tid >> 3;          // 0..31 (x rows, 4 passes of 32)
  const int ac4  = (tid & 7) * 4;     // float4 column
  const int brow = tid >> 2;          // 0..63 (Keff rows, 2 passes of 64)
  const int bkc  = (tid & 3) * 8;     // 16-byte K chunk

  float4 xr[4];  // in-flight x tile

  auto load_x = [&](int k0) {
#pragma unroll
    for (int p = 0; p < 4; ++p)
      xr[p] = *(const float4*)&X[(size_t)(m0 + arow + p * 32) * DDIM + k0 + ac4];
  };
  auto issue_async_B = [&](int k0, int buf) {
#pragma unroll
    for (int p = 0; p < 2; ++p) {
      const int n = brow + p * 64;
      async_copy_b128((unsigned)(uintptr_t)&Bhs[buf][n * BK + bkc],
                      &Bh[(size_t)(n0 + n) * DDIM + k0 + bkc]);
      async_copy_b128((unsigned)(uintptr_t)&Bls[buf][n * BK + bkc],
                      &Bl[(size_t)(n0 + n) * DDIM + k0 + bkc]);
    }
  };
  auto split_store_A = [&](int buf) {
#pragma unroll
    for (int p = 0; p < 4; ++p) {
      const int r = arow + p * 32;
      const float fv[4] = {xr[p].x, xr[p].y, xr[p].z, xr[p].w};
      union { bf16_t h[4]; v2u u; } uh, ul;
#pragma unroll
      for (int q = 0; q < 4; ++q) {
        const bf16_t h = (bf16_t)fv[q];
        uh.h[q] = h;
        ul.h[q] = (bf16_t)(fv[q] - (float)h);
      }
      *(v2u*)&Ah[buf][r * BK + ac4] = uh.u;
      *(v2u*)&Al[buf][r * BK + ac4] = ul.u;
    }
  };

  // ---- prologue: stage tile 0 into buffer 0 ----
  load_x(0);
  issue_async_B(0, 0);
  split_store_A(0);
  wait_asynccnt0();
  __syncthreads();

  const int NT = DDIM / BK;  // 128 K-steps
  for (int kt = 0; kt < NT; ++kt) {
    const int cur = kt & 1;
    const int nxt = cur ^ 1;
    const bool more = (kt + 1) < NT;

    // issue next tile's traffic before touching this tile's compute
    if (more) {
      load_x((kt + 1) * BK);
      issue_async_B((kt + 1) * BK, nxt);
    }

    // ---- fragments from LDS per documented 16-bit WMMA striping ----
    v16bf afh[4], afl[4], bfh[2], bfl[2];
    const int kc0 = half * 8;
#pragma unroll
    for (int ms = 0; ms < 4; ++ms) {
      const int row = waveM * 64 + ms * 16 + l16;
      const v8bf h0 = *(const v8bf*)&Ah[cur][row * BK + kc0];
      const v8bf h1 = *(const v8bf*)&Ah[cur][row * BK + kc0 + 16];
      afh[ms] = __builtin_shufflevector(h0, h1, 0, 1, 2, 3, 4, 5, 6, 7,
                                        8, 9, 10, 11, 12, 13, 14, 15);
      const v8bf l0 = *(const v8bf*)&Al[cur][row * BK + kc0];
      const v8bf l1 = *(const v8bf*)&Al[cur][row * BK + kc0 + 16];
      afl[ms] = __builtin_shufflevector(l0, l1, 0, 1, 2, 3, 4, 5, 6, 7,
                                        8, 9, 10, 11, 12, 13, 14, 15);
    }
#pragma unroll
    for (int ns = 0; ns < 2; ++ns) {
      const int col = waveN * 32 + ns * 16 + l16;
      const v8bf h0 = *(const v8bf*)&Bhs[cur][col * BK + kc0];
      const v8bf h1 = *(const v8bf*)&Bhs[cur][col * BK + kc0 + 16];
      bfh[ns] = __builtin_shufflevector(h0, h1, 0, 1, 2, 3, 4, 5, 6, 7,
                                        8, 9, 10, 11, 12, 13, 14, 15);
      const v8bf l0 = *(const v8bf*)&Bls[cur][col * BK + kc0];
      const v8bf l1 = *(const v8bf*)&Bls[cur][col * BK + kc0 + 16];
      bfl[ns] = __builtin_shufflevector(l0, l1, 0, 1, 2, 3, 4, 5, 6, 7,
                                        8, 9, 10, 11, 12, 13, 14, 15);
    }

    // ---- 4x2 subtiles x 3 split-products = 24 WMMAs ----
#pragma unroll
    for (int ms = 0; ms < 4; ++ms) {
#pragma unroll
      for (int ns = 0; ns < 2; ++ns) {
        acc[ms][ns] = __builtin_amdgcn_wmma_f32_16x16x32_bf16(
            false, afh[ms], false, bfh[ns], (short)0, acc[ms][ns], false, false);
        acc[ms][ns] = __builtin_amdgcn_wmma_f32_16x16x32_bf16(
            false, afh[ms], false, bfl[ns], (short)0, acc[ms][ns], false, false);
        acc[ms][ns] = __builtin_amdgcn_wmma_f32_16x16x32_bf16(
            false, afl[ms], false, bfh[ns], (short)0, acc[ms][ns], false, false);
      }
    }

    // ---- finish staging next buffer, then one barrier per iteration ----
    if (more) split_store_A(nxt);
    wait_asynccnt0();
    __syncthreads();
  }

  // ---- epilogue: + bias, store per C-layout (VGPR r -> M=r / r+8) ----
#pragma unroll
  for (int ns = 0; ns < 2; ++ns) {
    const int ng = n0 + waveN * 32 + ns * 16 + l16;
    const float bv = bias[ng];
#pragma unroll
    for (int ms = 0; ms < 4; ++ms) {
      const int mbase = m0 + waveM * 64 + ms * 16 + half * 8;
#pragma unroll
      for (int r = 0; r < 8; ++r) {
        out[(size_t)(mbase + r) * DDIM + ng] = acc[ms][ns][r] + bv;
      }
    }
  }
}

// ---------------------------------------------------------------------------
extern "C" void kernel_launch(void* const* d_in, const int* in_sizes, int n_in,
                              void* d_out, int out_size, void* d_ws, size_t ws_size,
                              hipStream_t stream) {
  const float* x    = (const float*)d_in[0];
  const float* W    = (const float*)d_in[1];
  const float* bias = (const float*)d_in[2];
  const float* a0   = (const float*)d_in[3];
  const float* b0   = (const float*)d_in[4];
  const float* a1   = (const float*)d_in[5];
  const float* b1   = (const float*)d_in[6];
  float* out = (float*)d_out;

  // workspace: Keff split hi/lo, transposed [N][K], bf16 -> 2 * 32 MB
  bf16_t* Bh = (bf16_t*)d_ws;
  bf16_t* Bl = Bh + (size_t)DDIM * DDIM;

  const int M = in_sizes[0] / DDIM;  // 8 * 2048 = 16384

  loha_prep<<<dim3(DDIM / 256, DDIM), 256, 0, stream>>>(W, a0, b0, a1, b1, Bh, Bl);
  loha_gemm<<<dim3(DDIM / BN, M / BM), 256, 0, stream>>>(x, Bh, Bl, bias, out);
}
